// My_RNN_26603027431714
// MI455X (gfx1250) — matrix operand order repaired and checked by reading, not compiled, not run on hardware
//
#include <hip/hip_runtime.h>
#include <hip/hip_bf16.h>
#include <math.h>

// ---------------------------------------------------------------------------
// Types for WMMA fragments (trivially copyable ext vectors for bit_cast)
// ---------------------------------------------------------------------------
typedef __attribute__((ext_vector_type(16))) __bf16 v16bf;
typedef __attribute__((ext_vector_type(8)))  float  v8f;
typedef unsigned int u32;
typedef unsigned short u16;
typedef unsigned long long u64;
typedef __attribute__((ext_vector_type(4))) u32 u32x4;

struct BFPack { u32x4 lo; u32x4 hi; };

__device__ __forceinline__ v16bf to_v16bf(u32x4 lo, u32x4 hi) {
    BFPack t{lo, hi};
    return __builtin_bit_cast(v16bf, t);
}

__device__ __forceinline__ u16 f2bf(float f) {
    u32 u = __builtin_bit_cast(u32, f);
    u32 r = u + 0x7FFFu + ((u >> 16) & 1u);   // round-to-nearest-even
    return (u16)(r >> 16);
}

__device__ __forceinline__ float sigmoidf_(float x) {
    return 1.0f / (1.0f + expf(-x));
}

// LDS byte address = low 32 bits of the flat shared pointer (ISA 10.2:
// LDS aperture)
__device__ __forceinline__ u32 lds_addr32(const void* p) {
    return (u32)(unsigned long long)(uintptr_t)p;
}

#define CDIV(a, b) (((a) + (b) - 1) / (b))

// ---------------------------------------------------------------------------
// Core bf16 WMMA GEMM:  C[M][N] = A[M][K] @ Bt^T   (Bt is [N][K] bf16)
//   mode 0: C = acc + addm[M][N]          (LSTM preactivation, addm may alias C)
//   mode 1: C = relu(acc + bias[N])
//   mode 2: C = acc + bias[N]
// Block tile 128x128, BK=32, 8 waves of 64x32, v_wmma_f32_16x16x32_bf16.
// Tile staging uses global_load_async_to_lds_b128 (ASYNCcnt) double-buffered.
// ---------------------------------------------------------------------------
__global__ __launch_bounds__(256) void k_gemm_bf16(
    const u16* __restrict__ A,    // [M][K] bf16 row-major
    const u16* __restrict__ Bt,   // [N][K] bf16 row-major (pre-transposed W)
    float* __restrict__ C,        // [M][N] f32
    const float* __restrict__ bias,
    const float* __restrict__ addm,
    int M, int N, int K, int mode)
{
    __shared__ u16 As[2][128][32];   // [m][k]  8KB * 2
    __shared__ u16 Bs[2][128][32];   // [n][k]  8KB * 2

    const int tid  = threadIdx.x;
    const int lane = tid & 31;
    const int wave = tid >> 5;
    const int wm   = wave >> 2;   // 0..1  -> 64-row slab
    const int wn   = wave & 3;    // 0..3  -> 32-col slab
    const int bm0  = blockIdx.x * 128;
    const int bn0  = blockIdx.y * 128;

    v8f acc[4][2];
    const v8f vzero = {0.f, 0.f, 0.f, 0.f, 0.f, 0.f, 0.f, 0.f};
#pragma unroll
    for (int i = 0; i < 4; ++i)
#pragma unroll
        for (int j = 0; j < 2; ++j) acc[i][j] = vzero;

    // Issue async cache->LDS copies for one 128x32 A tile and 128x32 B^T tile.
    // 512 16B chunks per tile, 2 per thread per tile.
    auto load_tiles_async = [&](int kt, int buf) {
        const int k0 = kt * 32;
#pragma unroll
        for (int c = 0; c < 2; ++c) {
            int ch  = tid * 2 + c;
            int row = ch >> 2;          // 0..127
            int kc  = (ch & 3) * 8;     // 0,8,16,24
            u32 la = lds_addr32(&As[buf][row][kc]);
            u64 ga = (u64)(uintptr_t)(A + (size_t)(bm0 + row) * K + k0 + kc);
            asm volatile("global_load_async_to_lds_b128 %0, %1, off"
                         :: "v"(la), "v"(ga) : "memory");
            u32 lb = lds_addr32(&Bs[buf][row][kc]);
            u64 gb = (u64)(uintptr_t)(Bt + (size_t)(bn0 + row) * K + k0 + kc);
            asm volatile("global_load_async_to_lds_b128 %0, %1, off"
                         :: "v"(lb), "v"(gb) : "memory");
        }
    };

    auto compute = [&](int buf) {
        const int ml = lane & 15;
        const int kh = lane >> 4;   // K-half select
        v16bf afr[4], bfr[2];
        // A fragments: 16-bit A 16x32 layout (lanes 0-15: K 0-7 & 16-23,
        // lanes 16-31: K 8-15 & 24-31)
#pragma unroll
        for (int i = 0; i < 4; ++i) {
            int row = wm * 64 + i * 16 + ml;
            u32x4 lo = *reinterpret_cast<const u32x4*>(&As[buf][row][kh * 8]);
            u32x4 hi = *reinterpret_cast<const u32x4*>(&As[buf][row][16 + kh * 8]);
            afr[i] = to_v16bf(lo, hi);
        }
        // B fragments: lanes 0-15 hold K 0-7 & 16-23, lanes 16-31 hold K 8-15 & 24-31
#pragma unroll
        for (int j = 0; j < 2; ++j) {
            int col = wn * 32 + j * 16 + ml;
            u32x4 lo = *reinterpret_cast<const u32x4*>(&Bs[buf][col][kh * 8]);
            u32x4 hi = *reinterpret_cast<const u32x4*>(&Bs[buf][col][16 + kh * 8]);
            bfr[j] = to_v16bf(lo, hi);
        }
#pragma unroll
        for (int i = 0; i < 4; ++i)
#pragma unroll
            for (int j = 0; j < 2; ++j)
                acc[i][j] = __builtin_amdgcn_wmma_f32_16x16x32_bf16(
                    false, afr[i], false, bfr[j], (short)0, acc[i][j],
                    false, false);
    };

    const int nk = K / 32;
    load_tiles_async(0, 0);
    asm volatile("s_wait_asynccnt 0" ::: "memory");
    __syncthreads();
    int buf = 0;
    for (int kt = 0; kt < nk; ++kt) {
        if (kt + 2 < nk) {  // global_prefetch_b8 of the tile after next
            __builtin_prefetch(A + (size_t)(bm0 + (tid >> 1)) * K + (kt + 2) * 32, 0, 0);
            __builtin_prefetch(Bt + (size_t)(bn0 + (tid >> 1)) * K + (kt + 2) * 32, 0, 0);
        }
        if (kt + 1 < nk) load_tiles_async(kt + 1, buf ^ 1);  // overlaps compute
        compute(buf);
        asm volatile("s_wait_asynccnt 0" ::: "memory");
        __syncthreads();
        buf ^= 1;
    }

    // Epilogue. C/D layout: lanes 0-15 N=lane, M=r; lanes 16-31 N=lane-16, M=8+r
    const int ml = lane & 15;
    const int rh = (lane >> 4) * 8;
#pragma unroll
    for (int i = 0; i < 4; ++i) {
#pragma unroll
        for (int j = 0; j < 2; ++j) {
            int gn = bn0 + wn * 32 + j * 16 + ml;
            int gm = bm0 + wm * 64 + i * 16 + rh;
#pragma unroll
            for (int r = 0; r < 8; ++r) {
                size_t off = (size_t)(gm + r) * N + gn;
                float v = acc[i][j][r];
                if (mode == 0) v += addm[off];
                else           v += bias[gn];
                if (mode == 1 && v < 0.f) v = 0.f;
                C[off] = v;
            }
        }
    }
}

// ---------------------------------------------------------------------------
// Elementwise / reduction kernels
// ---------------------------------------------------------------------------
__global__ void k_fill_f32(float* p, float v, int n) {
    int i = blockIdx.x * blockDim.x + threadIdx.x;
    if (i < n) p[i] = v;
}
__global__ void k_fill_u16(u16* p, u16 v, int n) {
    int i = blockIdx.x * blockDim.x + threadIdx.x;
    if (i < n) p[i] = v;
}

// Transposed convert: W[K][N] f32 -> out[N][K] bf16
__global__ void k_cvt_bf16_t(const float* __restrict__ W, u16* __restrict__ out,
                             int N, int K) {
    int idx = blockIdx.x * blockDim.x + threadIdx.x;
    if (idx >= N * K) return;
    int n = idx / K, k = idx - n * K;
    out[idx] = f2bf(W[(size_t)k * N + n]);
}

// Uall^T[2048][512] bf16, gate order f,i,o,c: out[g][k] = U{gate}[k][j]
__global__ void k_pack_Ut(const float* U0, const float* U1, const float* U2,
                          const float* U3, u16* out, int n) {
    int idx = blockIdx.x * blockDim.x + threadIdx.x;
    if (idx >= n) return;
    int g = idx >> 9, k = idx & 511;
    int gate = g >> 9, j = g & 511;
    const float* U = (gate == 0) ? U0 : (gate == 1) ? U1 : (gate == 2) ? U2 : U3;
    out[idx] = f2bf(U[k * 512 + j]);
}
// Aall[3][2048] f32
__global__ void k_pack_A(const float* A0, const float* A1, const float* A2,
                         const float* A3, float* out, int n) {
    int idx = blockIdx.x * blockDim.x + threadIdx.x;
    if (idx >= n) return;
    int r = idx >> 11, g = idx & 2047;
    int gate = g >> 9, j = g & 511;
    const float* A = (gate == 0) ? A0 : (gate == 1) ? A1 : (gate == 2) ? A2 : A3;
    out[idx] = A[r * 512 + j];
}
// biasAU[2048] = A_b + U_b per gate
__global__ void k_pack_bias(const float* Ab0, const float* Ab1, const float* Ab2,
                            const float* Ab3, const float* Ub0, const float* Ub1,
                            const float* Ub2, const float* Ub3, float* out) {
    int g = blockIdx.x * blockDim.x + threadIdx.x;
    if (g >= 2048) return;
    int gate = g >> 9, j = g & 511;
    const float* Ab = (gate == 0) ? Ab0 : (gate == 1) ? Ab1 : (gate == 2) ? Ab2 : Ab3;
    const float* Ub = (gate == 0) ? Ub0 : (gate == 1) ? Ub1 : (gate == 2) ? Ub2 : Ub3;
    out[g] = Ab[j] + Ub[j];
}

// Steering vectors a_r, a_i [B][64]
__global__ void k_steer(const float* __restrict__ phi, float* __restrict__ ar,
                        float* __restrict__ ai, int n) {
    int idx = blockIdx.x * blockDim.x + threadIdx.x;
    if (idx >= n) return;
    int b = idx >> 6, m = idx & 63;
    float arg = 3.14159265358979323846f * sinf(phi[b]) * (float)m;
    ar[idx] = cosf(arg);
    ai[idx] = sinf(arg);
}

// xa[B][2048] = x @ Aall + (A_b + U_b), x = [y0, y1, snr]
__global__ void k_xa(const float* __restrict__ y, const float* __restrict__ Aall,
                     const float* __restrict__ biasAU, const int* __restrict__ Pp,
                     float* __restrict__ xa, int n) {
    int idx = blockIdx.x * blockDim.x + threadIdx.x;
    if (idx >= n) return;
    int b = idx >> 11, g = idx & 2047;
    float P = (float)Pp[0];
    float snr = (log10f(P) - 1.0f) * rsqrtf(1.6666f);
    float x0 = y[b * 2 + 0], x1 = y[b * 2 + 1];
    xa[idx] = x0 * Aall[g] + x1 * Aall[2048 + g] + snr * Aall[4096 + g] + biasAU[g];
}

// LSTM pointwise: gpre[B][2048] -> updates c[B][512], writes h as bf16
__global__ void k_lstm(const float* __restrict__ gpre, float* __restrict__ c,
                       u16* __restrict__ hbf, int n) {
    int idx = blockIdx.x * blockDim.x + threadIdx.x;
    if (idx >= n) return;
    int b = idx >> 9, j = idx & 511;
    const float* gp = gpre + (size_t)b * 2048;
    float f = sigmoidf_(gp[j]);
    float i = sigmoidf_(gp[512 + j]);
    float o = sigmoidf_(gp[1024 + j]);
    float g = tanhf(gp[1536 + j]);
    float cn = f * c[idx] + i * g;
    c[idx] = cn;
    hbf[idx] = f2bf(o * tanhf(cn));
}

// Per-column mean/var over B rows (coalesced: one thread per column)
__global__ void k_colstats(const float* __restrict__ X, float* __restrict__ mean,
                           float* __restrict__ var, int B, int N) {
    int n = blockIdx.x * blockDim.x + threadIdx.x;
    if (n >= N) return;
    float s = 0.f, q = 0.f;
    for (int r = 0; r < B; ++r) {
        float v = X[(size_t)r * N + n];
        s += v; q += v * v;
    }
    float mu = s / (float)B;
    mean[n] = mu;
    var[n] = q / (float)B - mu * mu;
}

__global__ void k_bn(const float* __restrict__ X, const float* __restrict__ mean,
                     const float* __restrict__ var, const float* __restrict__ g,
                     const float* __restrict__ bb, u16* __restrict__ out,
                     int n, int N) {
    int idx = blockIdx.x * blockDim.x + threadIdx.x;
    if (idx >= n) return;
    int c = idx % N;
    float v = (X[idx] - mean[c]) * rsqrtf(var[c] + 1e-5f) * g[c] + bb[c];
    out[idx] = f2bf(v);
}

// Beamforming: normalize w[b][128], combine with steering, emit y[b][2]
__global__ __launch_bounds__(64) void k_beam(
    const float* __restrict__ w, const float* __restrict__ ar,
    const float* __restrict__ ai, const float* __restrict__ alr,
    const float* __restrict__ ali, const float* __restrict__ nr,
    const float* __restrict__ ni, const int* __restrict__ Pp,
    float* __restrict__ y) {
    int b = blockIdx.x;
    int m = threadIdx.x;  // 0..63
    __shared__ float sh0[64], sh1[64], sh2[64];
    float wr = w[(size_t)b * 128 + m];
    float wi = w[(size_t)b * 128 + 64 + m];
    sh0[m] = wr * wr + wi * wi;
    __syncthreads();
    for (int st = 32; st > 0; st >>= 1) {
        if (m < st) sh0[m] += sh0[m + st];
        __syncthreads();
    }
    float inv = rsqrtf(sh0[0]);
    __syncthreads();
    wr *= inv; wi *= inv;
    float cr = ar[(size_t)b * 64 + m], ci = ai[(size_t)b * 64 + m];
    sh1[m] = wr * cr - wi * ci;
    sh2[m] = wr * ci + wi * cr;
    __syncthreads();
    for (int st = 32; st > 0; st >>= 1) {
        if (m < st) { sh1[m] += sh1[m + st]; sh2[m] += sh2[m + st]; }
        __syncthreads();
    }
    if (m == 0) {
        float yr = sh1[0], yi = sh2[0];
        float sp = sqrtf((float)Pp[0]);
        float a = alr[b], bI = ali[b];
        const float NOISE_STD = 0.70710678118654752f;
        float ycr = sp * (yr * a - yi * bI) + NOISE_STD * nr[b];
        float yci = sp * (yr * bI + yi * a) + NOISE_STD * ni[b];
        y[b * 2 + 0] = ycr / sp;
        y[b * 2 + 1] = yci / sp;
    }
}

// phi_hat = c @ phi_w + phi_b ; err[b] = (phi[b]-phi_hat)^2
__global__ __launch_bounds__(128) void k_phierr(
    const float* __restrict__ c, const float* __restrict__ pw,
    const float* __restrict__ pb, const float* __restrict__ phi,
    float* __restrict__ err) {
    int b = blockIdx.x;
    float s = 0.f;
    for (int j = threadIdx.x; j < 512; j += 128)
        s += c[(size_t)b * 512 + j] * pw[j];
    __shared__ float sh[128];
    sh[threadIdx.x] = s;
    __syncthreads();
    for (int st = 64; st > 0; st >>= 1) {
        if (threadIdx.x < st) sh[threadIdx.x] += sh[threadIdx.x + st];
        __syncthreads();
    }
    if (threadIdx.x == 0) {
        float ph = sh[0] + pb[0];
        float d = phi[b] - ph;
        err[b] = d * d;
    }
}

__global__ __launch_bounds__(256) void k_loss(const float* __restrict__ err,
                                              float* __restrict__ out, int B) {
    float s = 0.f;
    for (int i = threadIdx.x; i < B; i += 256) s += err[i];
    __shared__ float sh[256];
    sh[threadIdx.x] = s;
    __syncthreads();
    for (int st = 128; st > 0; st >>= 1) {
        if (threadIdx.x < st) sh[threadIdx.x] += sh[threadIdx.x + st];
        __syncthreads();
    }
    if (threadIdx.x == 0) out[0] = sh[0] / (float)B;
}

// ---------------------------------------------------------------------------
// Host-side orchestration
// ---------------------------------------------------------------------------
extern "C" void kernel_launch(void* const* d_in, const int* in_sizes, int n_in,
                              void* d_out, int out_size, void* d_ws, size_t ws_size,
                              hipStream_t stream) {
    const float* alr = (const float*)d_in[0];
    const float* ali = (const float*)d_in[1];
    const float* phi = (const float*)d_in[2];
    const float* nrl = (const float*)d_in[3];
    const float* nil = (const float*)d_in[4];
    const float* Afw = (const float*)d_in[5];  const float* Afb = (const float*)d_in[6];
    const float* Aiw = (const float*)d_in[7];  const float* Aib = (const float*)d_in[8];
    const float* Aow = (const float*)d_in[9];  const float* Aob = (const float*)d_in[10];
    const float* Acw = (const float*)d_in[11]; const float* Acb = (const float*)d_in[12];
    const float* Ufw = (const float*)d_in[13]; const float* Ufb = (const float*)d_in[14];
    const float* Uiw = (const float*)d_in[15]; const float* Uib = (const float*)d_in[16];
    const float* Uow = (const float*)d_in[17]; const float* Uob = (const float*)d_in[18];
    const float* Ucw = (const float*)d_in[19]; const float* Ucb = (const float*)d_in[20];
    const float* f1w = (const float*)d_in[21]; const float* f1b = (const float*)d_in[22];
    const float* f2w = (const float*)d_in[23]; const float* f2b = (const float*)d_in[24];
    const float* f3w = (const float*)d_in[25]; const float* f3b = (const float*)d_in[26];
    const float* f4w = (const float*)d_in[27]; const float* f4b = (const float*)d_in[28];
    const float* b1g = (const float*)d_in[29]; const float* b1b = (const float*)d_in[30];
    const float* b2g = (const float*)d_in[31]; const float* b2b = (const float*)d_in[32];
    const float* b3g = (const float*)d_in[33]; const float* b3b = (const float*)d_in[34];
    const float* pw  = (const float*)d_in[35]; const float* pb  = (const float*)d_in[36];
    const int*   Pp  = (const int*)d_in[39];

    const int B   = in_sizes[0];           // 8192
    const int TAU = in_sizes[3] / B;       // 10

    // Workspace arena
    char* base = (char*)d_ws;
    size_t off = 0;
    auto alloc = [&](size_t bytes) -> void* {
        void* p = base + off;
        off = (off + bytes + 255) & ~(size_t)255;
        return p;
    };
    u16*   Uallt  = (u16*)alloc((size_t)2048 * 512 * 2);   // [2048][512]
    float* Aall   = (float*)alloc((size_t)3 * 2048 * 4);
    float* biasAU = (float*)alloc((size_t)2048 * 4);
    u16*   w1t    = (u16*)alloc((size_t)1024 * 512 * 2);   // [1024][512]
    u16*   w2t    = (u16*)alloc((size_t)1024 * 1024 * 2);
    u16*   w3t    = (u16*)alloc((size_t)1024 * 1024 * 2);
    u16*   w4t    = (u16*)alloc((size_t)128 * 1024 * 2);   // [128][1024]
    float* ar     = (float*)alloc((size_t)B * 64 * 4);
    float* ai     = (float*)alloc((size_t)B * 64 * 4);
    float* xa     = (float*)alloc((size_t)B * 2048 * 4);
    u16*   hbf    = (u16*)alloc((size_t)B * 512 * 2);
    float* cst    = (float*)alloc((size_t)B * 512 * 4);
    float* actf   = (float*)alloc((size_t)B * 1024 * 4);
    u16*   actbf  = (u16*)alloc((size_t)B * 1024 * 2);
    float* wbuf   = (float*)alloc((size_t)B * 128 * 4);
    float* yreal  = (float*)alloc((size_t)B * 2 * 4);
    float* meanb  = (float*)alloc((size_t)1024 * 4);
    float* varb   = (float*)alloc((size_t)1024 * 4);
    float* err    = (float*)alloc((size_t)B * 4);

    auto gemm = [&](const u16* A, const u16* Wt, float* C, const float* bias,
                    const float* addm, int M, int N, int K, int mode) {
        dim3 grid(M / 128, N / 128);
        k_gemm_bf16<<<grid, 256, 0, stream>>>(A, Wt, C, bias, addm, M, N, K, mode);
    };

    // ---- one-time (per launch) weight prep + steering + state init ----
    k_pack_Ut<<<CDIV(2048 * 512, 256), 256, 0, stream>>>(Ufw, Uiw, Uow, Ucw, Uallt, 2048 * 512);
    k_pack_A<<<CDIV(3 * 2048, 256), 256, 0, stream>>>(Afw, Aiw, Aow, Acw, Aall, 3 * 2048);
    k_pack_bias<<<CDIV(2048, 256), 256, 0, stream>>>(Afb, Aib, Aob, Acb, Ufb, Uib, Uob, Ucb, biasAU);
    k_cvt_bf16_t<<<CDIV(1024 * 512, 256), 256, 0, stream>>>(f1w, w1t, 1024, 512);
    k_cvt_bf16_t<<<CDIV(1024 * 1024, 256), 256, 0, stream>>>(f2w, w2t, 1024, 1024);
    k_cvt_bf16_t<<<CDIV(1024 * 1024, 256), 256, 0, stream>>>(f3w, w3t, 1024, 1024);
    k_cvt_bf16_t<<<CDIV(128 * 1024, 256), 256, 0, stream>>>(f4w, w4t, 128, 1024);
    k_steer<<<CDIV(B * 64, 256), 256, 0, stream>>>(phi, ar, ai, B * 64);
    k_fill_u16<<<CDIV(B * 512, 256), 256, 0, stream>>>(hbf, (u16)0, B * 512);
    k_fill_f32<<<CDIV(B * 512, 256), 256, 0, stream>>>(cst, 0.0f, B * 512);
    k_fill_f32<<<CDIV(B * 2, 256), 256, 0, stream>>>(yreal, 1.0f, B * 2);

    // ---- tau scan steps ----
    for (int t = 0; t < TAU; ++t) {
        // LSTM
        k_xa<<<CDIV(B * 2048, 256), 256, 0, stream>>>(yreal, Aall, biasAU, Pp, xa, B * 2048);
        gemm(hbf, Uallt, xa, nullptr, xa, B, 2048, 512, 0);     // xa += h @ Uall
        k_lstm<<<CDIV(B * 512, 256), 256, 0, stream>>>(xa, cst, hbf, B * 512);
        // MLP
        gemm(hbf, w1t, actf, f1b, nullptr, B, 1024, 512, 1);
        k_colstats<<<CDIV(1024, 256), 256, 0, stream>>>(actf, meanb, varb, B, 1024);
        k_bn<<<CDIV(B * 1024, 256), 256, 0, stream>>>(actf, meanb, varb, b1g, b1b, actbf, B * 1024, 1024);
        gemm(actbf, w2t, actf, f2b, nullptr, B, 1024, 1024, 1);
        k_colstats<<<CDIV(1024, 256), 256, 0, stream>>>(actf, meanb, varb, B, 1024);
        k_bn<<<CDIV(B * 1024, 256), 256, 0, stream>>>(actf, meanb, varb, b2g, b2b, actbf, B * 1024, 1024);
        gemm(actbf, w3t, actf, f3b, nullptr, B, 1024, 1024, 1);
        k_colstats<<<CDIV(1024, 256), 256, 0, stream>>>(actf, meanb, varb, B, 1024);
        k_bn<<<CDIV(B * 1024, 256), 256, 0, stream>>>(actf, meanb, varb, b3g, b3b, actbf, B * 1024, 1024);
        gemm(actbf, w4t, wbuf, f4b, nullptr, B, 128, 1024, 2);
        // Beamform + channel
        k_beam<<<B, 64, 0, stream>>>(wbuf, ar, ai, alr, ali,
                                     nrl + (size_t)t * B, nil + (size_t)t * B, Pp, yreal);
    }

    // ---- final LSTM + loss ----
    k_xa<<<CDIV(B * 2048, 256), 256, 0, stream>>>(yreal, Aall, biasAU, Pp, xa, B * 2048);
    gemm(hbf, Uallt, xa, nullptr, xa, B, 2048, 512, 0);
    k_lstm<<<CDIV(B * 512, 256), 256, 0, stream>>>(xa, cst, hbf, B * 512);
    k_phierr<<<B, 128, 0, stream>>>(cst, pw, pb, phi, err);
    k_loss<<<1, 256, 0, stream>>>(err, (float*)d_out, B);

    (void)n_in; (void)out_size; (void)ws_size;
}
